// GINConvLayer_24163486007673
// MI455X (gfx1250) — matrix operand order repaired and checked by reading, not compiled, not run on hardware
//
#include <hip/hip_runtime.h>

// GIN conv layer for MI455X (gfx1250): wave32 + bf16 WMMA GEMMs.
// Algebraic rewrite: segsum(efeat@We + be, dst) = segsum(efeat,dst)@We + deg*be,
// so the 800K-row edge GEMM collapses to a 50K-row GEMM after scatter-add.

#define D   128
#define TILE_M 16
#define WPB 5                    // waves per block (50000/16 = 3125 = 5*625 tiles)
#define GEMM_BLOCK (WPB * 32)
#define BN_EPS 1e-5f
#define WSTRIDE 20               // dwords per (kb,n) group: 16 data + 4 pad
#define WL_WORDS (4 * D * WSTRIDE)   // 40 KB

typedef __attribute__((ext_vector_type(16))) __bf16 v16bf;
typedef __attribute__((ext_vector_type(2)))  __bf16 v2bf;
typedef __attribute__((ext_vector_type(16))) float  v16f;
typedef __attribute__((ext_vector_type(8)))  float  v8f;

union BfVec  { v16bf v; unsigned u[8]; uint4 q[2]; };
union F32Vec { v16f f; float4 q[4]; };

// f32 pair -> packed 2x bf16 dword; let ISel pick a packed cvt if available
__device__ __forceinline__ unsigned pack2bf(float lo, float hi) {
  v2bf p = { (__bf16)lo, (__bf16)hi };
  return __builtin_bit_cast(unsigned, p);
}

// 16 f32 -> 16 bf16 in WMMA A-operand order, as one vector conversion
__device__ __forceinline__ v16bf packA(float4 a0, float4 a1, float4 a2, float4 a3) {
  F32Vec af;
  af.q[0] = a0; af.q[1] = a1; af.q[2] = a2; af.q[3] = a3;
  return __builtin_convertvector(af.f, v16bf);
}

// Fill LDS weight buffer: group (kb, n) holds the 16 packed K-pair dwords a
// lane needs contiguously -> B operand = two ds_load_b128.
// WSTRIDE=20 keeps groups 16B-aligned and spreads lanes across banks.
__device__ __forceinline__ void fill_wlds(unsigned* wl, const float* __restrict__ W,
                                          int tid) {
  for (int i = tid; i < 4 * D * 16; i += GEMM_BLOCK) {
    int kb  = i >> 11;          // / (D*16)
    int rem = i & 2047;
    int n   = rem >> 4;
    int kl  = rem & 15;
    int k2  = kb * 16 + kl;     // bf16-pair index: covers K = 2*k2, 2*k2+1
    wl[(kb * D + n) * WSTRIDE + kl] =
        pack2bf(W[(2 * k2) * D + n], W[(2 * k2 + 1) * D + n]);
  }
}

__device__ __forceinline__ v16bf load_b(const unsigned* wl, int kb, int n, int sub) {
  BfVec B;
  const unsigned* bp = wl + ((kb * D + n) * WSTRIDE + sub * 4);
  B.q[0] = *(const uint4*)bp;
  B.q[1] = *(const uint4*)(bp + 8);
  return B.v;
}

__device__ __forceinline__ float4 bnrelu4(float4 a, float4 s, float4 t) {
  float4 r;
  r.x = fmaxf(fmaf(a.x, s.x, t.x), 0.0f);
  r.y = fmaxf(fmaf(a.y, s.y, t.y), 0.0f);
  r.z = fmaxf(fmaf(a.z, s.z, t.z), 0.0f);
  r.w = fmaxf(fmaf(a.w, s.w, t.w), 0.0f);
  return r;
}

// ---------------- K0: init workspace ----------------
__global__ void k_init(const float* __restrict__ nfeat, float* __restrict__ xbuf,
                       float* __restrict__ eagg, float* __restrict__ deg,
                       float* __restrict__ gstats, int nnodes) {
  size_t i = (size_t)blockIdx.x * blockDim.x + threadIdx.x;
  size_t nd = (size_t)nnodes * D;
  if (i < nd)             { xbuf[i] = nfeat[i]; eagg[i] = 0.0f; }
  if (i < (size_t)nnodes) deg[i] = 0.0f;
  if (i < 2 * D)          gstats[i] = 0.0f;
}

// ---------------- K1: edge gather/scatter (one wave per edge) ----------------
__global__ void k_edge_scatter(const float* __restrict__ nfeat,
                               const float* __restrict__ efeat,
                               const int* __restrict__ src,
                               const int* __restrict__ dst,
                               float* __restrict__ xbuf,  // accumulates nfeat[src]
                               float* __restrict__ eagg,  // accumulates efeat
                               float* __restrict__ deg, int nedges) {
  int widx = (int)(((size_t)blockIdx.x * blockDim.x + threadIdx.x) >> 5);
  int lane = threadIdx.x & 31;
  if (widx >= nedges) return;
  int s = src[widx], d = dst[widx];
  float4 nf = *(const float4*)(nfeat + (size_t)s * D + lane * 4);
  float4 ef = *(const float4*)(efeat + (size_t)widx * D + lane * 4);
  float* xp = xbuf + (size_t)d * D + lane * 4;
  float* ep = eagg + (size_t)d * D + lane * 4;
  atomicAdd(xp + 0, nf.x); atomicAdd(xp + 1, nf.y);
  atomicAdd(xp + 2, nf.z); atomicAdd(xp + 3, nf.w);
  atomicAdd(ep + 0, ef.x); atomicAdd(ep + 1, ef.y);
  atomicAdd(ep + 2, ef.z); atomicAdd(ep + 3, ef.w);
  if (lane == 0) atomicAdd(deg + d, 1.0f);
}

// ---------------- K2: h1 = (nfeat+agg)@W1 + b1, + BN batch stats ----------------
__global__ __launch_bounds__(GEMM_BLOCK) void k_gemm1_bn_stats(
    const float* __restrict__ x, const float* __restrict__ W1,
    const float* __restrict__ b1, float* __restrict__ h1,
    float* __restrict__ gstats, int nrows) {
  __shared__ unsigned wl[WL_WORDS];    // 40 KB
  __shared__ float sstat[2 * D];
  const int tid = threadIdx.x;
  fill_wlds(wl, W1, tid);
  for (int i = tid; i < 2 * D; i += GEMM_BLOCK) sstat[i] = 0.0f;
  __syncthreads();

  const int lane = tid & 31;
  const int sub  = lane >> 4;     // 0: K 0-7,16-23, 1: K 8-15,24-31 (ISA A/B layout)
  const int lm   = lane & 15;     // row (A) / column (B,C) within tile
  const int m0   = (blockIdx.x * WPB + (tid >> 5)) * TILE_M;

  if (m0 < nrows) {               // nrows assumed multiple of 16 (50000 = 3125*16)
    const float* arow = x + (size_t)(m0 + lm) * D;
    v8f acc[8];
#pragma unroll
    for (int nt = 0; nt < 8; ++nt) acc[nt] = v8f{};

#pragma unroll
    for (int kb = 0; kb < 4; ++kb) {
      const int c0 = kb * 32 + sub * 8;
      v16bf A = packA(*(const float4*)(arow + c0),
                      *(const float4*)(arow + c0 + 4),
                      *(const float4*)(arow + c0 + 16),
                      *(const float4*)(arow + c0 + 20));
#pragma unroll
      for (int nt = 0; nt < 8; ++nt) {
        v16bf B = load_b(wl, kb, nt * 16 + lm, sub);
        acc[nt] = __builtin_amdgcn_wmma_f32_16x16x32_bf16(
            false, A, false, B, (short)0, acc[nt], false, false);
      }
    }

    const int rowbase = m0 + sub * 8;   // C layout: VGPR r -> row rowbase+r
#pragma unroll
    for (int nt = 0; nt < 8; ++nt) {
      const int n = nt * 16 + lm;
      const float bias = b1[n];
      float* orow = h1 + (size_t)rowbase * D + n;
      float s1 = 0.0f, s2 = 0.0f;
#pragma unroll
      for (int r = 0; r < 8; ++r) {
        float v = acc[nt][r] + bias;
        orow[(size_t)r * D] = v;
        s1 += v; s2 += v * v;
      }
      atomicAdd(&sstat[n], s1);
      atomicAdd(&sstat[D + n], s2);
    }
  }
  __syncthreads();
  for (int i = tid; i < 2 * D; i += GEMM_BLOCK) atomicAdd(&gstats[i], sstat[i]);
}

// ---------------- K3: fold BN stats into scale/shift ----------------
__global__ void k_bn_finalize(const float* __restrict__ gstats,
                              const float* __restrict__ gamma,
                              const float* __restrict__ beta,
                              float* __restrict__ bnp, int nnodes) {
  int c = threadIdx.x;
  if (c < D) {
    float inv_n = 1.0f / (float)nnodes;
    float mu  = gstats[c] * inv_n;
    float var = gstats[D + c] * inv_n - mu * mu;
    float s   = gamma[c] * rsqrtf(var + BN_EPS);
    bnp[c]     = s;
    bnp[D + c] = beta[c] - mu * s;
  }
}

// ---------------- K4: out = relu(bn(h1))@W2 + eagg@We + b2 + deg*be ----------------
// Both GEMMs accumulate into the SAME WMMA accumulator; LDS holds one bf16
// weight matrix at a time (refilled between phases).
__global__ __launch_bounds__(GEMM_BLOCK) void k_gemm2_fused(
    const float* __restrict__ h1, const float* __restrict__ eagg,
    const float* __restrict__ deg, const float* __restrict__ bnp,
    const float* __restrict__ W2, const float* __restrict__ b2,
    const float* __restrict__ We, const float* __restrict__ be,
    float* __restrict__ out, int nrows) {
  __shared__ unsigned wl[WL_WORDS];    // 40 KB, reused for W2 then We
  __shared__ float sc[D], sh[D];
  const int tid = threadIdx.x;
  for (int i = tid; i < D; i += GEMM_BLOCK) { sc[i] = bnp[i]; sh[i] = bnp[D + i]; }
  fill_wlds(wl, W2, tid);
  __syncthreads();

  const int lane = tid & 31;
  const int sub  = lane >> 4;
  const int lm   = lane & 15;
  const int m0   = (blockIdx.x * WPB + (tid >> 5)) * TILE_M;
  const bool active = (m0 < nrows);

  v8f acc[8];
#pragma unroll
  for (int nt = 0; nt < 8; ++nt) acc[nt] = v8f{};

  // ---- phase 1: A = relu(bn(h1)), B = W2 ----
  if (active) {
    const float* arow = h1 + (size_t)(m0 + lm) * D;
#pragma unroll
    for (int kb = 0; kb < 4; ++kb) {
      const int c0 = kb * 32 + sub * 8;
      v16bf A = packA(
          bnrelu4(*(const float4*)(arow + c0),
                  *(const float4*)(sc + c0),      *(const float4*)(sh + c0)),
          bnrelu4(*(const float4*)(arow + c0 + 4),
                  *(const float4*)(sc + c0 + 4),  *(const float4*)(sh + c0 + 4)),
          bnrelu4(*(const float4*)(arow + c0 + 16),
                  *(const float4*)(sc + c0 + 16), *(const float4*)(sh + c0 + 16)),
          bnrelu4(*(const float4*)(arow + c0 + 20),
                  *(const float4*)(sc + c0 + 20), *(const float4*)(sh + c0 + 20)));
#pragma unroll
      for (int nt = 0; nt < 8; ++nt) {
        v16bf B = load_b(wl, kb, nt * 16 + lm, sub);
        acc[nt] = __builtin_amdgcn_wmma_f32_16x16x32_bf16(
            false, A, false, B, (short)0, acc[nt], false, false);
      }
    }
  }

  // ---- swap weights: W2 -> We ----
  __syncthreads();
  fill_wlds(wl, We, tid);
  __syncthreads();

  // ---- phase 2: A = eagg, B = We, accumulate into same C; epilogue ----
  if (active) {
    const float* arow = eagg + (size_t)(m0 + lm) * D;
#pragma unroll
    for (int kb = 0; kb < 4; ++kb) {
      const int c0 = kb * 32 + sub * 8;
      v16bf A = packA(*(const float4*)(arow + c0),
                      *(const float4*)(arow + c0 + 4),
                      *(const float4*)(arow + c0 + 16),
                      *(const float4*)(arow + c0 + 20));
#pragma unroll
      for (int nt = 0; nt < 8; ++nt) {
        v16bf B = load_b(wl, kb, nt * 16 + lm, sub);
        acc[nt] = __builtin_amdgcn_wmma_f32_16x16x32_bf16(
            false, A, false, B, (short)0, acc[nt], false, false);
      }
    }

    const int rowbase = m0 + sub * 8;
    float4 dg0 = *(const float4*)(deg + rowbase);
    float4 dg1 = *(const float4*)(deg + rowbase + 4);
    float dg[8] = {dg0.x, dg0.y, dg0.z, dg0.w, dg1.x, dg1.y, dg1.z, dg1.w};
#pragma unroll
    for (int nt = 0; nt < 8; ++nt) {
      const int n = nt * 16 + lm;
      const float bb2 = b2[n];
      const float bbe = be[n];
      float* orow = out + (size_t)rowbase * D + n;
#pragma unroll
      for (int r = 0; r < 8; ++r)
        orow[(size_t)r * D] = acc[nt][r] + bb2 + dg[r] * bbe;
    }
  }
}

// ---------------- host ----------------
extern "C" void kernel_launch(void* const* d_in, const int* in_sizes, int n_in,
                              void* d_out, int out_size, void* d_ws, size_t ws_size,
                              hipStream_t stream) {
  const float* nfeat = (const float*)d_in[0];
  const float* efeat = (const float*)d_in[1];
  const int*   src   = (const int*)d_in[2];
  const int*   dst   = (const int*)d_in[3];
  const float* W1    = (const float*)d_in[4];
  const float* b1    = (const float*)d_in[5];
  const float* gamma = (const float*)d_in[6];
  const float* beta  = (const float*)d_in[7];
  const float* W2    = (const float*)d_in[8];
  const float* b2    = (const float*)d_in[9];
  const float* We    = (const float*)d_in[10];
  const float* be    = (const float*)d_in[11];
  float* out = (float*)d_out;

  const int nnodes = in_sizes[0] / D;     // 50000
  const int nedges = in_sizes[2];         // 800000
  const size_t nd = (size_t)nnodes * D;

  // workspace layout (floats): xbuf | eagg | h1 | deg | gstats | bnp  (~77 MB)
  float* xbuf   = (float*)d_ws;
  float* eagg   = xbuf + nd;
  float* h1     = eagg + nd;
  float* deg    = h1 + nd;
  float* gstats = deg + nnodes;
  float* bnp    = gstats + 2 * D;

  {
    int blocks = (int)((nd + 255) / 256);
    k_init<<<blocks, 256, 0, stream>>>(nfeat, xbuf, eagg, deg, gstats, nnodes);
  }
  {
    int blocks = (nedges + 7) / 8;        // 8 waves (edges) per 256-thread block
    k_edge_scatter<<<blocks, 256, 0, stream>>>(nfeat, efeat, src, dst,
                                               xbuf, eagg, deg, nedges);
  }
  {
    int tiles  = (nnodes + TILE_M - 1) / TILE_M;   // 3125
    int blocks = (tiles + WPB - 1) / WPB;          // 625
    k_gemm1_bn_stats<<<blocks, GEMM_BLOCK, 0, stream>>>(xbuf, W1, b1, h1,
                                                        gstats, nnodes);
  }
  k_bn_finalize<<<1, D, 0, stream>>>(gstats, gamma, beta, bnp, nnodes);
  {
    int tiles  = (nnodes + TILE_M - 1) / TILE_M;
    int blocks = (tiles + WPB - 1) / WPB;
    k_gemm2_fused<<<blocks, GEMM_BLOCK, 0, stream>>>(h1, eagg, deg, bnp,
                                                     W2, b2, We, be, out, nnodes);
  }
}
